// BitLinearBitBLAS_3307124818587
// MI455X (gfx1250) — 8-bit matrix operand order repaired and checked by reading, compiled, not run on hardware
//
#include <hip/hip_runtime.h>
#include <hip/hip_fp16.h>
#include <stdint.h>

typedef __attribute__((ext_vector_type(8))) int   v8i;
typedef __attribute__((ext_vector_type(4))) float v4f;

#define M_DIM   16384
#define K_DIM   2048
#define N_DIM   2048
#define W_ELEMS (2048 * 2048)

#define WS_PARTIALS 0
#define WS_WSUM     4096
#define WS_SI       8192
#define WS_QW       131072
#define WS_QX       (131072 + 4194304)

__global__ __launch_bounds__(256) void k_wabs_partial(const float* __restrict__ w,
                                                      float* __restrict__ partials) {
    const int t = threadIdx.x;
    const float4* wp = (const float4*)w + (size_t)blockIdx.x * 1024;
    float s = 0.0f;
#pragma unroll
    for (int i = 0; i < 4; ++i) {
        float4 v = wp[i * 256 + t];
        s += fabsf(v.x) + fabsf(v.y) + fabsf(v.z) + fabsf(v.w);
    }
#pragma unroll
    for (int off = 16; off > 0; off >>= 1) s += __shfl_xor(s, off, 32);
    __shared__ float red[8];
    if ((t & 31) == 0) red[t >> 5] = s;
    __syncthreads();
    if (t == 0) {
        float tot = 0.0f;
#pragma unroll
        for (int i = 0; i < 8; ++i) tot += red[i];
        partials[blockIdx.x] = tot;
    }
}

__global__ __launch_bounds__(256) void k_wabs_final(const float* __restrict__ partials,
                                                    float* __restrict__ wsum) {
    const int t = threadIdx.x;
    float s = partials[t] + partials[t + 256] + partials[t + 512] + partials[t + 768];
#pragma unroll
    for (int off = 16; off > 0; off >>= 1) s += __shfl_xor(s, off, 32);
    __shared__ float red[8];
    if ((t & 31) == 0) red[t >> 5] = s;
    __syncthreads();
    if (t == 0) {
        float tot = 0.0f;
#pragma unroll
        for (int i = 0; i < 8; ++i) tot += red[i];
        *wsum = tot;
    }
}

__global__ __launch_bounds__(256) void k_wquant(const float* __restrict__ w,
                                                const float* __restrict__ wsum,
                                                uint32_t* __restrict__ qw_packed) {
    const float mean = *wsum * (1.0f / (float)W_ELEMS);
    const float sw = 1.0f / fmaxf(mean, 1e-5f);
    const size_t idx = (size_t)blockIdx.x * 256 + threadIdx.x;
    float4 v = ((const float4*)w)[idx];
    int q0 = (int)fminf(fmaxf(rintf(v.x * sw), -1.0f), 1.0f);
    int q1 = (int)fminf(fmaxf(rintf(v.y * sw), -1.0f), 1.0f);
    int q2 = (int)fminf(fmaxf(rintf(v.z * sw), -1.0f), 1.0f);
    int q3 = (int)fminf(fmaxf(rintf(v.w * sw), -1.0f), 1.0f);
    qw_packed[idx] = (uint32_t)(q0 & 0xff) | ((uint32_t)(q1 & 0xff) << 8) |
                     ((uint32_t)(q2 & 0xff) << 16) | ((uint32_t)(q3 & 0xff) << 24);
}

__global__ __launch_bounds__(256) void k_actquant(const float* __restrict__ x,
                                                  uint32_t* __restrict__ qx_packed,
                                                  float* __restrict__ si) {
    const int row = blockIdx.x;
    const int t = threadIdx.x;
    const v4f* xr = (const v4f*)(x + (size_t)row * K_DIM);
    v4f v0 = __builtin_nontemporal_load(xr + t * 2);
    v4f v1 = __builtin_nontemporal_load(xr + t * 2 + 1);
    float m = fmaxf(fmaxf(fmaxf(fabsf(v0[0]), fabsf(v0[1])), fmaxf(fabsf(v0[2]), fabsf(v0[3]))),
                    fmaxf(fmaxf(fabsf(v1[0]), fabsf(v1[1])), fmaxf(fabsf(v1[2]), fabsf(v1[3]))));
#pragma unroll
    for (int off = 16; off > 0; off >>= 1) m = fmaxf(m, __shfl_xor(m, off, 32));
    __shared__ float red[8];
    if ((t & 31) == 0) red[t >> 5] = m;
    __syncthreads();
    float amax = red[0];
#pragma unroll
    for (int i = 1; i < 8; ++i) amax = fmaxf(amax, red[i]);
    const float s = 127.0f / fmaxf(amax, 1e-5f);
    if (t == 0) si[row] = s;
    int q[8];
#pragma unroll
    for (int j = 0; j < 4; ++j) {
        q[j]     = (int)fminf(fmaxf(rintf(v0[j] * s), -128.0f), 127.0f);
        q[j + 4] = (int)fminf(fmaxf(rintf(v1[j] * s), -128.0f), 127.0f);
    }
    uint32_t p0 = (uint32_t)(q[0] & 0xff) | ((uint32_t)(q[1] & 0xff) << 8) |
                  ((uint32_t)(q[2] & 0xff) << 16) | ((uint32_t)(q[3] & 0xff) << 24);
    uint32_t p1 = (uint32_t)(q[4] & 0xff) | ((uint32_t)(q[5] & 0xff) << 8) |
                  ((uint32_t)(q[6] & 0xff) << 16) | ((uint32_t)(q[7] & 0xff) << 24);
    uint32_t* dst = qx_packed + (size_t)row * (K_DIM / 4) + t * 2;
    dst[0] = p0;
    dst[1] = p1;
}

__global__ __launch_bounds__(256) void k_gemm(const int8_t* __restrict__ qx,
                                              const int8_t* __restrict__ qw,
                                              const float* __restrict__ si,
                                              const float* __restrict__ wsum,
                                              _Float16* __restrict__ out) {
    const int tid  = threadIdx.x;
    const int wave = tid >> 5;
    const int lane = tid & 31;
    const int l16  = lane & 15;
    const int lhi  = lane >> 4;

    const int m0 = blockIdx.y * 256 + wave * 32;
    const int n0 = blockIdx.x * 64;

    const int8_t* aptr0 = qx + (size_t)(m0 + l16) * K_DIM + lhi * 8;
    const int8_t* aptr1 = aptr0 + (size_t)16 * K_DIM;
    const int8_t* bbase = qw + (size_t)(n0 + l16) * K_DIM + lhi * 8;

    v8i acc[2][4];
#pragma unroll
    for (int s = 0; s < 2; ++s)
#pragma unroll
        for (int t4 = 0; t4 < 4; ++t4) acc[s][t4] = v8i{};

#pragma unroll 1
    for (int k0 = 0; k0 < K_DIM; k0 += 64) {
        __builtin_prefetch(aptr0 + k0 + 64, 0, 3);
        __builtin_prefetch(aptr1 + k0 + 64, 0, 3);
        __builtin_prefetch(bbase + k0 + 64, 0, 3);

        uint2 a00 = *(const uint2*)(aptr0 + k0);
        uint2 a01 = *(const uint2*)(aptr0 + k0 + 16);
        uint2 a02 = *(const uint2*)(aptr0 + k0 + 32);
        uint2 a03 = *(const uint2*)(aptr0 + k0 + 48);
        uint2 a10 = *(const uint2*)(aptr1 + k0);
        uint2 a11 = *(const uint2*)(aptr1 + k0 + 16);
        uint2 a12 = *(const uint2*)(aptr1 + k0 + 32);
        uint2 a13 = *(const uint2*)(aptr1 + k0 + 48);
        v8i A0, A1;
        A0[0] = (int)a00.x; A0[1] = (int)a00.y; A0[2] = (int)a01.x; A0[3] = (int)a01.y;
        A0[4] = (int)a02.x; A0[5] = (int)a02.y; A0[6] = (int)a03.x; A0[7] = (int)a03.y;
        A1[0] = (int)a10.x; A1[1] = (int)a10.y; A1[2] = (int)a11.x; A1[3] = (int)a11.y;
        A1[4] = (int)a12.x; A1[5] = (int)a12.y; A1[6] = (int)a13.x; A1[7] = (int)a13.y;

#pragma unroll
        for (int t4 = 0; t4 < 4; ++t4) {
            const int8_t* bp = bbase + (size_t)(16 * t4) * K_DIM + k0;
            uint2 b0 = *(const uint2*)bp;
            uint2 b1 = *(const uint2*)(bp + 16);
            uint2 b2 = *(const uint2*)(bp + 32);
            uint2 b3 = *(const uint2*)(bp + 48);
            v8i B;
            B[0] = (int)b0.x; B[1] = (int)b0.y; B[2] = (int)b1.x; B[3] = (int)b1.y;
            B[4] = (int)b2.x; B[5] = (int)b2.y; B[6] = (int)b3.x; B[7] = (int)b3.y;
            acc[0][t4] = __builtin_amdgcn_wmma_i32_16x16x64_iu8(true, A0, true, B,
                                                                acc[0][t4], false, false);
            acc[1][t4] = __builtin_amdgcn_wmma_i32_16x16x64_iu8(true, A1, true, B,
                                                                acc[1][t4], false, false);
        }
    }

    const float mean = *wsum * (1.0f / (float)W_ELEMS);
    const float sw = 1.0f / fmaxf(mean, 1e-5f);

#pragma unroll
    for (int s = 0; s < 2; ++s) {
        const int mbase = m0 + s * 16 + lhi * 8;
        float si_r[8];
#pragma unroll
        for (int r = 0; r < 8; ++r) si_r[r] = si[mbase + r];
#pragma unroll
        for (int t4 = 0; t4 < 4; ++t4) {
            const int ncol = n0 + 16 * t4 + l16;
#pragma unroll
            for (int r = 0; r < 8; ++r) {
                float v = ((float)acc[s][t4][r] / si_r[r]) / sw;
                __builtin_nontemporal_store((_Float16)v,
                                            out + (size_t)(mbase + r) * N_DIM + ncol);
            }
        }
    }
}

extern "C" void kernel_launch(void* const* d_in, const int* in_sizes, int n_in,
                              void* d_out, int out_size, void* d_ws, size_t ws_size,
                              hipStream_t stream) {
    const float* x = (const float*)d_in[0];
    const float* w = (const float*)d_in[1];

    char* ws = (char*)d_ws;
    float*    partials = (float*)(ws + WS_PARTIALS);
    float*    wsum     = (float*)(ws + WS_WSUM);
    float*    si       = (float*)(ws + WS_SI);
    int8_t*   qw       = (int8_t*)(ws + WS_QW);
    int8_t*   qx       = (int8_t*)(ws + WS_QX);
    _Float16* out      = (_Float16*)d_out;

    k_wabs_partial<<<1024, 256, 0, stream>>>(w, partials);
    k_wabs_final<<<1, 256, 0, stream>>>(partials, wsum);
    k_wquant<<<W_ELEMS / (256 * 4), 256, 0, stream>>>(w, wsum, (uint32_t*)qw);
    k_actquant<<<M_DIM, 256, 0, stream>>>(x, (uint32_t*)qx, si);

    dim3 grid(N_DIM / 64, M_DIM / 256);
    k_gemm<<<grid, 256, 0, stream>>>(qx, qw, si, wsum, out);
}
